// SurvivalAttentionLayer_9268539425503
// MI455X (gfx1250) — matrix operand order repaired and checked by reading, not compiled, not run on hardware
//
#include <hip/hip_runtime.h>
#include <hip/hip_bf16.h>
#include <cstdint>
#include <cstddef>

// ---------------------------------------------------------------------------
// SurvivalAttentionLayer on MI455X (gfx1250), wave32 + WMMA f32_16x16x32_f16.
// ~58 GFLOP total; all intermediates stream through L2 (192MB); S x S score
// matrices are never materialized (flash-attention over 32-key chunks).
// ---------------------------------------------------------------------------

typedef __attribute__((ext_vector_type(16))) _Float16 v16h;
typedef __attribute__((ext_vector_type(8)))  _Float16 v8h;
typedef __attribute__((ext_vector_type(4)))  _Float16 v4h;
typedef __attribute__((ext_vector_type(8)))  float    v8f;
typedef __attribute__((ext_vector_type(4)))  float    v4f;

#define SHUF16(lo,hi) __builtin_shufflevector((lo),(hi),0,1,2,3,4,5,6,7,8,9,10,11,12,13,14,15)
#define WMMA(A,B,C) __builtin_amdgcn_wmma_f32_16x16x32_f16(false,(A),false,(B),(short)0,(C),false,false)

// ------------------------------ f32 -> f16 convert -------------------------
__global__ __launch_bounds__(256)
void k_cvt_f16(const float* __restrict__ s, _Float16* __restrict__ d) {
  size_t i = ((size_t)blockIdx.x * 256 + threadIdx.x) * 4;
  v4f v = *(const v4f*)(s + i);
  v4h o;
  o[0] = (_Float16)v[0]; o[1] = (_Float16)v[1];
  o[2] = (_Float16)v[2]; o[3] = (_Float16)v[3];
  *(v4h*)(d + i) = o;
}

// ------------------------------ WMMA GEMM ----------------------------------
// C[M,N] = A[M,K](f16) x W[N,K]^T(f16) + bias, compile-time ReLU / f32-out.
// Block = 256 threads = 8 waves; block tile 128x128; wave tile 64x32.
// A-fragment: lane m<16 holds K {0..7,16..23}; lane m>=16 holds {8..15,24..31}.
// B-fragment: lane n<16 holds K {0..7,16..23}; lane n>=16 holds {8..15,24..31}.
template <bool RELU, bool F32OUT>
__global__ __launch_bounds__(256)
void k_gemm(const _Float16* __restrict__ A, int K,
            const _Float16* __restrict__ W,
            const float* __restrict__ bias,
            void* __restrict__ Cout, int N)
{
  const int lane = threadIdx.x & 31, wave = threadIdx.x >> 5;
  const int ln = lane & 15, lh = lane >> 4;
  const int m0 = blockIdx.y * 128 + (wave >> 2) * 64;
  const int n0 = blockIdx.x * 128 + (wave & 3) * 32;
  v8f acc[4][2] = {};
  for (int kk = 0; kk < K; kk += 32) {
    v16h af[4], bf[2];
#pragma unroll
    for (int mi = 0; mi < 4; ++mi) {
      const _Float16* p = A + (size_t)(m0 + mi * 16 + ln) * K + kk + lh * 8;
      v8h lo = *(const v8h*)p, hi = *(const v8h*)(p + 16);
      af[mi] = SHUF16(lo, hi);
    }
#pragma unroll
    for (int ni = 0; ni < 2; ++ni) {
      const _Float16* p = W + (size_t)(n0 + ni * 16 + ln) * K + kk + lh * 8;
      v8h lo = *(const v8h*)p, hi = *(const v8h*)(p + 16);
      bf[ni] = SHUF16(lo, hi);
    }
#pragma unroll
    for (int mi = 0; mi < 4; ++mi)
#pragma unroll
      for (int ni = 0; ni < 2; ++ni)
        acc[mi][ni] = WMMA(af[mi], bf[ni], acc[mi][ni]);
  }
  // C tile layout: vgpr r, lane l -> row = r + 8*(l/16), col = l%16
#pragma unroll
  for (int ni = 0; ni < 2; ++ni) {
    const int col = n0 + ni * 16 + ln;
    const float bv = bias[col];
#pragma unroll
    for (int mi = 0; mi < 4; ++mi) {
      const int row0 = m0 + mi * 16 + lh * 8;
#pragma unroll
      for (int r = 0; r < 8; ++r) {
        float v = acc[mi][ni][r] + bv;
        if constexpr (RELU) v = fmaxf(v, 0.f);
        if constexpr (F32OUT)
          __builtin_nontemporal_store(v, (float*)Cout + (size_t)(row0 + r) * N + col);
        else
          ((_Float16*)Cout)[(size_t)(row0 + r) * N + col] = (_Float16)v;
      }
    }
  }
}

// --------------------- LDS-tiled f16 transpose (for V^T) -------------------
// dst[n*rows + m] = src[m*ld + col0 + n]
__global__ __launch_bounds__(256)
void k_transpose_f16(const _Float16* __restrict__ src, int ld, int col0,
                     _Float16* __restrict__ dst, int rows)
{
  __shared__ _Float16 tile[32][33];
  const int m0 = blockIdx.x * 32, c0 = blockIdx.y * 32;
  const int tx = threadIdx.x, ty = threadIdx.y;  // 32 x 8
#pragma unroll
  for (int j = 0; j < 4; ++j)
    tile[ty + j * 8][tx] = src[(size_t)(m0 + ty + j * 8) * ld + col0 + c0 + tx];
  __syncthreads();
#pragma unroll
  for (int j = 0; j < 4; ++j)
    dst[(size_t)(c0 + ty + j * 8) * rows + m0 + tx] = tile[tx][ty + j * 8];
}

// ------------------------- flash attention (WMMA) --------------------------
// One wave owns a 16-query tile, streams keys in chunks of 32.
// Computes S^T = K x Q^T so softmax reduces in-lane + one shfl_xor(16), and
// P^T converts directly into the B-fragment of O^T = V^T x P^T.
// All chunk loads are issued up front from induction-variable pointers with
// immediate offsets so the scheduler can clause them and overlap with VALU.
#define AW 8
__global__ __launch_bounds__(32 * AW)
void k_flash_attn(const _Float16* __restrict__ Q, int qld,
                  const _Float16* __restrict__ Kp, int kld,
                  const _Float16* __restrict__ Vt,       // [nheads*64, BS]
                  const float* __restrict__ bias,        // [B*S] per-key additive
                  _Float16* __restrict__ O, int oldm,
                  int S, int BS, int nheads)
{
  const int lane = threadIdx.x & 31, wave = threadIdx.x >> 5;
  const int ln = lane & 15, lh = lane >> 4;
  const int b = blockIdx.y / nheads, h = blockIdx.y % nheads;
  const int q0 = (blockIdx.x * AW + wave) * 16;

  // Q as B-fragments: column n = query q0+ln; K-dim = 64 dims in two steps.
  const _Float16* qrow = Q + (size_t)(b * S + q0 + ln) * qld + h * 64;
  v16h qf[2];
#pragma unroll
  for (int ks = 0; ks < 2; ++ks) {
    const _Float16* p = qrow + ks * 32 + lh * 8;
    v8h lo = *(const v8h*)p, hi = *(const v8h*)(p + 16);
    qf[ks] = SHUF16(lo, hi);
  }

  // Induction-variable row pointers (advance by constants each chunk).
  const size_t kstep = (size_t)32 * kld;
  const _Float16* kr0 = Kp + (size_t)(b * S + ln) * kld + h * 64 + lh * 8;        // key tile 0
  const _Float16* kr1 = kr0 + (size_t)16 * kld;                                   // key tile 1
  const _Float16* vr[4];
#pragma unroll
  for (int dt = 0; dt < 4; ++dt)
    vr[dt] = Vt + (size_t)(h * 64 + dt * 16 + ln) * BS + (size_t)b * S + lh * 16;
  const float* bp = bias + (size_t)b * S + lh * 8;

  v8f o[4] = {};
  float m_run = -3.0e38f, l_run = 0.f;

  for (int kc = 0; kc < S; kc += 32) {
    // ---- issue ALL chunk loads first (immediate offsets off row pointers) --
    v8h klo[2][2], khi[2][2];           // [key tile][k-step]
#pragma unroll
    for (int ks = 0; ks < 2; ++ks) {
      klo[0][ks] = *(const v8h*)(kr0 + ks * 32);
      khi[0][ks] = *(const v8h*)(kr0 + ks * 32 + 16);
      klo[1][ks] = *(const v8h*)(kr1 + ks * 32);
      khi[1][ks] = *(const v8h*)(kr1 + ks * 32 + 16);
    }
    v8h vlo[4], vhi[4];
#pragma unroll
    for (int dt = 0; dt < 4; ++dt) {
      vlo[dt] = *(const v8h*)vr[dt];
      vhi[dt] = *(const v8h*)(vr[dt] + 8);
    }
    v4f b0[2], b1[2];
#pragma unroll
    for (int kt = 0; kt < 2; ++kt) {
      b0[kt] = *(const v4f*)(bp + kt * 16);
      b1[kt] = *(const v4f*)(bp + kt * 16 + 4);
    }
    kr0 += kstep; kr1 += kstep;
#pragma unroll
    for (int dt = 0; dt < 4; ++dt) vr[dt] += 32;
    bp += 32;

    // ---- S^T = K x Q^T -------------------------------------------------
    v8f st[2];
#pragma unroll
    for (int kt = 0; kt < 2; ++kt) {
      v8f s = {};
      s = WMMA(SHUF16(klo[kt][0], khi[kt][0]), qf[0], s);
      s = WMMA(SHUF16(klo[kt][1], khi[kt][1]), qf[1], s);
      st[kt] = s;
    }
    // scale 1/sqrt(64) + per-key bias; chunk max over 16 keys held in-lane
    float cmax = -3.0e38f;
#pragma unroll
    for (int kt = 0; kt < 2; ++kt)
#pragma unroll
      for (int r = 0; r < 8; ++r) {
        float v = st[kt][r] * 0.125f + (r < 4 ? b0[kt][r] : b1[kt][r - 4]);
        st[kt][r] = v;
        cmax = fmaxf(cmax, v);
      }
    cmax = fmaxf(cmax, __shfl_xor(cmax, 16, 32));
    float m_new = fmaxf(m_run, cmax);
    float rescale = __expf(m_run - m_new);  // 0 on first chunk
    float csum = 0.f;
#pragma unroll
    for (int kt = 0; kt < 2; ++kt)
#pragma unroll
      for (int r = 0; r < 8; ++r) {
        float pv = __expf(st[kt][r] - m_new);
        st[kt][r] = pv;
        csum += pv;
      }
    csum += __shfl_xor(csum, 16, 32);
    l_run = l_run * rescale + csum;
    m_run = m_new;
#pragma unroll
    for (int dt = 0; dt < 4; ++dt)
#pragma unroll
      for (int r = 0; r < 8; ++r) o[dt][r] *= rescale;

    // Pack P^T into B-fragment: lanes 0..15 need keys 0..15, lanes 16..31
    // need keys 16..31 -> one half-wave exchange, then f32->f16 pack.
    float ex0[8], ex1[8];
#pragma unroll
    for (int r = 0; r < 8; ++r) {
      ex0[r] = __shfl_xor(st[0][r], 16, 32);
      ex1[r] = __shfl_xor(st[1][r], 16, 32);
    }
    v16h pf;
#pragma unroll
    for (int r = 0; r < 8; ++r) {
      pf[r]     = (_Float16)(lh ? ex1[r] : st[0][r]);
      pf[r + 8] = (_Float16)(lh ? st[1][r] : ex0[r]);
    }
    // ---- O^T += V^T x P^T ----------------------------------------------
#pragma unroll
    for (int dt = 0; dt < 4; ++dt)
      o[dt] = WMMA(SHUF16(vlo[dt], vhi[dt]), pf, o[dt]);
  }
  const float inv = 1.f / l_run;
  _Float16* orow = O + (size_t)(b * S + q0 + ln) * oldm + h * 64 + lh * 8;
#pragma unroll
  for (int dt = 0; dt < 4; ++dt) {
    v8h ov;
#pragma unroll
    for (int r = 0; r < 8; ++r) ov[r] = (_Float16)(o[dt][r] * inv);
    *(v8h*)(orow + dt * 16) = ov;
  }
}

// -------------------- survival heads -> main-attn key bias -----------------
__global__ __launch_bounds__(256)
void k_survival_combine(const _Float16* __restrict__ an, const _Float16* __restrict__ am,
                        const _Float16* __restrict__ ad,
                        const float* __restrict__ nw2, const float* __restrict__ nb2,
                        const float* __restrict__ mw2, const float* __restrict__ mb2,
                        const float* __restrict__ dw2, const float* __restrict__ db2,
                        const int* __restrict__ mask, float* __restrict__ biasM, int BS)
{
  int t = blockIdx.x * 256 + threadIdx.x;
  if (t >= BS) return;
  float sn = nb2[0], sm = mb2[0], sd = db2[0];
  const _Float16* pn = an + (size_t)t * 128;
  const _Float16* pm = am + (size_t)t * 128;
  const _Float16* pd = ad + (size_t)t * 128;
  for (int i = 0; i < 128; ++i) {
    sn += (float)pn[i] * nw2[i];
    sm += (float)pm[i] * mw2[i];
    sd += (float)pd[i] * dw2[i];
  }
  float nval = (sn > 20.f) ? sn : log1pf(__expf(sn));   // softplus
  float mu = 1.f / (1.f + __expf(-sm));                 // sigmoid
  float delta = fmaxf(sd, 0.f);                         // relu
  float surv = logf(nval + 1e-8f) + logf(mu + 1e-8f) - delta;
  float mf = (float)mask[t];
  biasM[t] = 0.1f * surv + (1.f - mf) * (-1e9f);
}

__global__ __launch_bounds__(256)
void k_mask_bias(const int* __restrict__ mask, float* __restrict__ biasS, int BS)
{
  int t = blockIdx.x * 256 + threadIdx.x;
  if (t < BS) biasS[t] = (mask[t] > 0) ? 0.f : -1e9f;
}

// ---------------------------------------------------------------------------
extern "C" void kernel_launch(void* const* d_in, const int* in_sizes, int n_in,
                              void* d_out, int out_size, void* d_ws, size_t ws_size,
                              hipStream_t stream)
{
  (void)in_sizes; (void)n_in; (void)out_size; (void)ws_size;
  const int Bb = 2, S = 2048, H = 768, I = 256, BS = Bb * S;

  const float* x      = (const float*)d_in[0];
  const int*   mask   = (const int*)d_in[1];
  const float* qw = (const float*)d_in[2];  const float* qb = (const float*)d_in[3];
  const float* kw = (const float*)d_in[4];  const float* kb = (const float*)d_in[5];
  const float* vw = (const float*)d_in[6];  const float* vb = (const float*)d_in[7];
  const float* ow = (const float*)d_in[8];  const float* ob = (const float*)d_in[9];
  const float* sp_w = (const float*)d_in[10];   const float* sp_b = (const float*)d_in[11];
  const float* sa_in_w = (const float*)d_in[12]; const float* sa_in_b = (const float*)d_in[13];
  const float* sa_out_w = (const float*)d_in[14]; const float* sa_out_b = (const float*)d_in[15];
  const float* nw1 = (const float*)d_in[16]; const float* nb1 = (const float*)d_in[17];
  const float* nw2 = (const float*)d_in[18]; const float* nb2 = (const float*)d_in[19];
  const float* mw1 = (const float*)d_in[20]; const float* mb1 = (const float*)d_in[21];
  const float* mw2 = (const float*)d_in[22]; const float* mb2 = (const float*)d_in[23];
  const float* dw1 = (const float*)d_in[24]; const float* db1 = (const float*)d_in[25];
  const float* dw2 = (const float*)d_in[26]; const float* db2 = (const float*)d_in[27];

  // --- bump allocator on workspace (everything fits in L2 anyway) ---
  char* p = (char*)d_ws;
  auto alloc = [&](size_t bytes) -> char* {
    char* r = p; p += (bytes + 255) & ~(size_t)255; return r;
  };
  _Float16* x_h     = (_Float16*)alloc((size_t)BS * H * 2);
  _Float16* qw_h    = (_Float16*)alloc((size_t)H * H * 2);
  _Float16* kw_h    = (_Float16*)alloc((size_t)H * H * 2);
  _Float16* vw_h    = (_Float16*)alloc((size_t)H * H * 2);
  _Float16* ow_h    = (_Float16*)alloc((size_t)H * H * 2);
  _Float16* spw_h   = (_Float16*)alloc((size_t)I * H * 2);
  _Float16* sain_h  = (_Float16*)alloc((size_t)(3 * I) * I * 2);
  _Float16* saout_h = (_Float16*)alloc((size_t)I * I * 2);
  _Float16* nw1_h   = (_Float16*)alloc((size_t)128 * I * 2);
  _Float16* mw1_h   = (_Float16*)alloc((size_t)128 * I * 2);
  _Float16* dw1_h   = (_Float16*)alloc((size_t)128 * I * 2);
  _Float16* q_h     = (_Float16*)alloc((size_t)BS * H * 2);
  _Float16* k_h     = (_Float16*)alloc((size_t)BS * H * 2);
  _Float16* v_h     = (_Float16*)alloc((size_t)BS * H * 2);
  _Float16* vt_h    = (_Float16*)alloc((size_t)H * BS * 2);
  _Float16* h0_h    = (_Float16*)alloc((size_t)BS * I * 2);
  _Float16* qkv_s   = (_Float16*)alloc((size_t)BS * (3 * I) * 2);
  _Float16* vt_s    = (_Float16*)alloc((size_t)I * BS * 2);
  _Float16* sa_att  = (_Float16*)alloc((size_t)BS * I * 2);
  _Float16* hs_h    = (_Float16*)alloc((size_t)BS * I * 2);
  _Float16* an_h    = (_Float16*)alloc((size_t)BS * 128 * 2);
  _Float16* am_h    = (_Float16*)alloc((size_t)BS * 128 * 2);
  _Float16* ad_h    = (_Float16*)alloc((size_t)BS * 128 * 2);
  _Float16* att_out = (_Float16*)alloc((size_t)BS * H * 2);
  float*    bias_m  = (float*)alloc((size_t)BS * 4);
  float*    bias_s  = (float*)alloc((size_t)BS * 4);

  auto cvt = [&](const float* s, _Float16* d, size_t n) {
    k_cvt_f16<<<dim3((unsigned)(n / 1024)), 256, 0, stream>>>(s, d);
  };
  auto gemm = [&](const _Float16* A, const _Float16* W, const float* bias,
                  _Float16* C, int M, int N, int K) {
    k_gemm<false, false><<<dim3(N / 128, M / 128), 256, 0, stream>>>(A, K, W, bias, C, N);
  };
  auto gemm_relu = [&](const _Float16* A, const _Float16* W, const float* bias,
                       _Float16* C, int M, int N, int K) {
    k_gemm<true, false><<<dim3(N / 128, M / 128), 256, 0, stream>>>(A, K, W, bias, C, N);
  };
  auto gemm_f32 = [&](const _Float16* A, const _Float16* W, const float* bias,
                      float* C, int M, int N, int K) {
    k_gemm<false, true><<<dim3(N / 128, M / 128), 256, 0, stream>>>(A, K, W, bias, C, N);
  };

  // ---- f32 -> f16 converts ----
  cvt(x, x_h, (size_t)BS * H);
  cvt(qw, qw_h, (size_t)H * H);  cvt(kw, kw_h, (size_t)H * H);
  cvt(vw, vw_h, (size_t)H * H);  cvt(ow, ow_h, (size_t)H * H);
  cvt(sp_w, spw_h, (size_t)I * H);
  cvt(sa_in_w, sain_h, (size_t)3 * I * I);
  cvt(sa_out_w, saout_h, (size_t)I * I);
  cvt(nw1, nw1_h, (size_t)128 * I);
  cvt(mw1, mw1_h, (size_t)128 * I);
  cvt(dw1, dw1_h, (size_t)128 * I);
  k_mask_bias<<<dim3(BS / 256), 256, 0, stream>>>(mask, bias_s, BS);

  // ---- scorer path ----
  gemm(x_h, spw_h, sp_b, h0_h, BS, I, H);                        // h0 = x*sp_w^T
  gemm(h0_h, sain_h, sa_in_b, qkv_s, BS, 3 * I, I);              // qkv (q|k|v cols)
  k_transpose_f16<<<dim3(BS / 32, I / 32), dim3(32, 8), 0, stream>>>(qkv_s, 3 * I, 2 * I, vt_s, BS);
  k_flash_attn<<<dim3(S / (16 * AW), Bb * 4), 32 * AW, 0, stream>>>(
      qkv_s, 3 * I, qkv_s + I, 3 * I, vt_s, bias_s, sa_att, I, S, BS, 4);
  gemm(sa_att, saout_h, sa_out_b, hs_h, BS, I, I);               // out-proj
  gemm_relu(hs_h, nw1_h, nb1, an_h, BS, 128, I);                 // relu MLPs
  gemm_relu(hs_h, mw1_h, mb1, am_h, BS, 128, I);
  gemm_relu(hs_h, dw1_h, db1, ad_h, BS, 128, I);
  k_survival_combine<<<dim3(BS / 256), 256, 0, stream>>>(
      an_h, am_h, ad_h, nw2, nb2, mw2, mb2, dw2, db2, mask, bias_m, BS);

  // ---- main attention ----
  gemm(x_h, qw_h, qb, q_h, BS, H, H);
  gemm(x_h, kw_h, kb, k_h, BS, H, H);
  gemm(x_h, vw_h, vb, v_h, BS, H, H);
  k_transpose_f16<<<dim3(BS / 32, H / 32), dim3(32, 8), 0, stream>>>(v_h, H, 0, vt_h, BS);
  k_flash_attn<<<dim3(S / (16 * AW), Bb * 12), 32 * AW, 0, stream>>>(
      q_h, H, k_h, H, vt_h, bias_m, att_out, H, S, BS, 12);

  // ---- output projection (f32 out, non-temporal stores) ----
  gemm_f32(att_out, ow_h, ob, (float*)d_out, BS, H, H);
}
